// RelativeCrossAttentionLayer_79585743995177
// MI455X (gfx1250) — matrix operand order repaired and checked by reading, not compiled, not run on hardware
//
#include <hip/hip_runtime.h>
#include <stdint.h>

#define L_ 1024
#define S_ 2048
#define N_ 4
#define E_ 512
#define H_ 8
#define HD_ 64

typedef __attribute__((ext_vector_type(16))) __bf16 v16bf;
typedef __attribute__((ext_vector_type(8))) float v8f;

union ABFrag { v16bf v; uint4 u[2]; };

__device__ __forceinline__ unsigned short f2bf(float f) {
  unsigned int u = __float_as_uint(f);
  u += 0x7FFFu + ((u >> 16) & 1u);          // round-to-nearest-even
  return (unsigned short)(u >> 16);
}

__device__ __forceinline__ uint4 ldu4(const void* p) {
  return *reinterpret_cast<const uint4*>(p);
}

__device__ __forceinline__ v8f wmma_bf16(const ABFrag& a, const ABFrag& b, v8f c) {
  return __builtin_amdgcn_wmma_f32_16x16x32_bf16(false, a.v, false, b.v,
                                                 (short)0, c, false, false);
}

// Async global->LDS 16-byte copy (ASYNCcnt-tracked), per CDNA5 ISA.
__device__ __forceinline__ void async_ld16(unsigned lds_off,
                                           const unsigned short* g) {
  asm volatile("global_load_async_to_lds_b128 %0, %1, off"
               :: "v"(lds_off), "v"(g)
               : "memory");
}
__device__ __forceinline__ void wait_async() {
  asm volatile("s_wait_asynccnt 0x0" ::: "memory");
}
// Generic->LDS byte offset: low 32 bits of the flat LDS-aperture address.
__device__ __forceinline__ unsigned lds_off_of(const void* p) {
  return (unsigned)(uintptr_t)p;
}

// ---------------------------------------------------------------------------
// 16x16 output tile GEMM mainloop: C = A(row-major MxK, bf16) * W^T
// (W row-major Ncols x K, bf16), per the CDNA5 A/B fragment layouts.
// ---------------------------------------------------------------------------
template <int K>
__device__ __forceinline__ v8f gemm16x16(const unsigned short* A,
                                         const unsigned short* W,
                                         int row0, int col0) {
  const int lane = threadIdx.x;
  const int half = lane >> 4;
  const int lm = lane & 15;
  const unsigned short* aB = A + (size_t)(row0 + lm) * K + half * 8;
  const unsigned short* bB = W + (size_t)(col0 + lm) * K + half * 8;
  v8f c = {};
#pragma unroll 4
  for (int kb = 0; kb < K; kb += 32) {
    ABFrag a, b;
    a.u[0] = ldu4(aB + kb);
    a.u[1] = ldu4(aB + kb + 16);
    b.u[0] = ldu4(bB + kb);
    b.u[1] = ldu4(bB + kb + 16);
    c = wmma_bf16(a, b, c);
  }
  return c;
}

// ---------------------------------------------------------------------------
__global__ void k_convert(const float* __restrict__ src,
                          unsigned short* __restrict__ dst, int n) {
  int i = blockIdx.x * blockDim.x + threadIdx.x;
  if (i < n) dst[i] = f2bf(src[i]);
}

// q = (query @ Wq^T + bq) * scale, then rotary; output (N,H,L,HD) bf16
__global__ void k_qproj(const unsigned short* __restrict__ qbf,
                        const unsigned short* __restrict__ w,
                        const float* __restrict__ bias,
                        const float* __restrict__ qpos,
                        unsigned short* __restrict__ qrot) {
  const int tm = blockIdx.x;
  const int tn = blockIdx.y * blockDim.y + threadIdx.y;
  v8f c = gemm16x16<E_>(qbf, w, tm * 16, tn * 16);
  const int lane = threadIdx.x, half = lane >> 4, lm = lane & 15;
  const int col = tn * 16 + lm;
  const int h = col >> 6, hd = col & 63;
#pragma unroll
  for (int r = 0; r < 8; ++r) {
    int m = tm * 16 + r + half * 8;
    int l = m >> 2, n = m & 3;                 // rows are l*N + n, N=4
    float x = c[r] + bias[col];
    float xp = __shfl_xor(x, 1, 32);           // partner column (col^1)
    float xs = x * 0.125f, xps = xp * 0.125f;  // scale = HD^-0.5
    float x2 = (col & 1) ? xps : -xps;         // rotary pair term
    size_t pi = (((size_t)n * L_ + l) * E_ + col) * 2;
    float res = xs * qpos[pi] + x2 * qpos[pi + 1];
    qrot[(((size_t)n * H_ + h) * L_ + l) * HD_ + hd] = f2bf(res);
  }
}

// kv = value @ Wkv^T + bkv; K gets rotary -> (N,H,S,HD); V -> (N,H,HD,S)
__global__ void k_kvproj(const unsigned short* __restrict__ vbf,
                         const unsigned short* __restrict__ w,
                         const float* __restrict__ bias,
                         const float* __restrict__ vpos,
                         unsigned short* __restrict__ krot,
                         unsigned short* __restrict__ vt) {
  const int tm = blockIdx.x;
  const int tn = blockIdx.y * blockDim.y + threadIdx.y;
  v8f c = gemm16x16<E_>(vbf, w, tm * 16, tn * 16);
  const int lane = threadIdx.x, half = lane >> 4, lm = lane & 15;
  const int col = tn * 16 + lm;
#pragma unroll
  for (int r = 0; r < 8; ++r) {
    int m = tm * 16 + r + half * 8;
    int s = m >> 2, n = m & 3;
    float x = c[r] + bias[col];
    float xp = __shfl_xor(x, 1, 32);           // before any divergence
    if (col < E_) {                            // K path (tile-uniform branch)
      float x2 = (col & 1) ? xp : -xp;
      size_t pi = (((size_t)n * S_ + s) * E_ + col) * 2;
      float res = x * vpos[pi] + x2 * vpos[pi + 1];
      int h = col >> 6, hd = col & 63;
      krot[(((size_t)n * H_ + h) * S_ + s) * HD_ + hd] = f2bf(res);
    } else {                                   // V path, store transposed
      int e = col - E_;
      int h = e >> 6, hd = e & 63;
      vt[(((size_t)n * H_ + h) * HD_ + hd) * (size_t)S_ + s] = f2bf(x);
    }
  }
}

// ---------------------------------------------------------------------------
// Flash attention. Block = 4 waves sharing one (n,h); K/V tiles are staged
// into LDS with double-buffered GLOBAL_LOAD_ASYNC_TO_LDS_B128 so all four
// waves share one copy (4x less global traffic, transfer overlapped with
// WMMA compute). Each wave owns a 16-row Q tile and runs online softmax.
// ---------------------------------------------------------------------------
__global__ void __launch_bounds__(128)
k_attn(const unsigned short* __restrict__ qrot,
       const unsigned short* __restrict__ krot,
       const unsigned short* __restrict__ vt,
       unsigned short* __restrict__ aobf) {
  const int w = threadIdx.y;
  const int lane = threadIdx.x, half = lane >> 4, lm = lane & 15;
  const int tid = w * 32 + lane;
  const int nh = blockIdx.y;
  const int n = nh >> 3, h = nh & 7;
  const int l0 = (blockIdx.x * 4 + w) * 16;

  const unsigned short* qB = qrot + ((size_t)nh * L_ + l0) * HD_;
  const unsigned short* kB = krot + (size_t)nh * S_ * HD_;
  const unsigned short* vB = vt + (size_t)nh * HD_ * S_;

  // Block-shared K/V staging (double buffered) + per-wave softmax scratch.
  __shared__ __align__(16) unsigned short kTile[2][32][64];  // [s_local][hd]
  __shared__ __align__(16) unsigned short vTile[2][64][32];  // [hd][s_local]
  __shared__ __align__(16) float sscore[4][16][32];
  __shared__ __align__(16) unsigned short sptile[4][16][32];
  __shared__ float salpha[4][16];
  __shared__ float smax[4][16];
  __shared__ float ssum[4][16];
  __shared__ float sinv[4][16];

  const unsigned kOff[2] = {lds_off_of(&kTile[0][0][0]),
                            lds_off_of(&kTile[1][0][0])};
  const unsigned vOff[2] = {lds_off_of(&vTile[0][0][0]),
                            lds_off_of(&vTile[1][0][0])};

  // Q A-fragments for K-dim 0..31 and 32..63 (loaded once from global)
  ABFrag aq0, aq1;
  {
    const unsigned short* aB = qB + (size_t)lm * HD_ + half * 8;
    aq0.u[0] = ldu4(aB);      aq0.u[1] = ldu4(aB + 16);
    aq1.u[0] = ldu4(aB + 32); aq1.u[1] = ldu4(aB + 48);
  }

  v8f o0 = {}, o1 = {}, o2 = {}, o3 = {};
  if (lane < 16) { smax[w][lane] = -1e30f; ssum[w][lane] = 0.0f; }

  // Issue async copy of K/V tile `it` (32 S-columns) into buffer it&1.
  // 512 chunks of 16B total; 128 threads x 4 chunks.
  auto issueTile = [&](int it) {
    const int s0 = it * 32;
    const int buf = it & 1;
    const int c0 = tid, c1 = tid + 128;
    // K tile: contiguous 4KB at kB + s0*64
    async_ld16(kOff[buf] + c0 * 16, kB + (size_t)s0 * 64 + c0 * 8);
    async_ld16(kOff[buf] + c1 * 16, kB + (size_t)s0 * 64 + c1 * 8);
    // V tile: 64 rows (hd) of 64B from vB + hd*S + s0
    async_ld16(vOff[buf] + c0 * 16,
               vB + (size_t)(c0 >> 2) * S_ + s0 + (c0 & 3) * 8);
    async_ld16(vOff[buf] + c1 * 16,
               vB + (size_t)(c1 >> 2) * S_ + s0 + (c1 & 3) * 8);
  };

  issueTile(0);

  for (int it = 0; it < S_ / 32; ++it) {
    const int buf = it & 1;
    wait_async();        // this wave's async copies done
    __syncthreads();     // whole tile (all waves' chunks) visible
    if (it + 1 < S_ / 32) issueTile(it + 1);  // prefetch next tile

    const unsigned short* kt = &kTile[buf][0][0];
    const unsigned short* vtl = &vTile[buf][0][0];

    // scores S = Q(16x64) @ K^T : two 16-col tiles from LDS
    const unsigned short* ka = kt + (size_t)lm * 64 + half * 8;
    const unsigned short* kb2 = ka + 16 * 64;
    ABFrag b0a, b1a, b0b, b1b;
    b0a.u[0] = ldu4(ka);       b0a.u[1] = ldu4(ka + 16);
    b1a.u[0] = ldu4(ka + 32);  b1a.u[1] = ldu4(ka + 48);
    b0b.u[0] = ldu4(kb2);      b0b.u[1] = ldu4(kb2 + 16);
    b1b.u[0] = ldu4(kb2 + 32); b1b.u[1] = ldu4(kb2 + 48);
    v8f sa = {}, sb = {};
    sa = wmma_bf16(aq0, b0a, sa);
    sa = wmma_bf16(aq1, b1a, sa);
    sb = wmma_bf16(aq0, b0b, sb);
    sb = wmma_bf16(aq1, b1b, sb);

#pragma unroll
    for (int r = 0; r < 8; ++r) {
      sscore[w][r + half * 8][lm] = sa[r];
      sscore[w][r + half * 8][lm + 16] = sb[r];
    }

    // online softmax (one row per lane 0..15); wave-internal LDS is in-order
    if (lane < 16) {
      float mo = smax[w][lane];
      float mx = mo;
#pragma unroll
      for (int j = 0; j < 32; ++j) mx = fmaxf(mx, sscore[w][lane][j]);
      float sum = 0.0f;
#pragma unroll
      for (int j = 0; j < 32; ++j) {
        float p = __expf(sscore[w][lane][j] - mx);
        sum += p;
        sptile[w][lane][j] = f2bf(p);
      }
      float al = __expf(mo - mx);
      smax[w][lane] = mx;
      ssum[w][lane] = ssum[w][lane] * al + sum;
      salpha[w][lane] = al;
    }

#pragma unroll
    for (int r = 0; r < 8; ++r) {
      float al = salpha[w][r + half * 8];
      o0[r] *= al; o1[r] *= al; o2[r] *= al; o3[r] *= al;
    }

    // P as A-fragment (16x32 bf16) from LDS
    ABFrag ap;
    const unsigned short* pB = &sptile[w][lm][half * 8];
    ap.u[0] = ldu4(pB);
    ap.u[1] = ldu4(pB + 16);

    // O += P @ V : V tile is (hd, s_local) so B-fragments are contiguous
    ABFrag bv;
    const unsigned short* v0 = vtl + (size_t)lm * 32 + half * 8;
    bv.u[0] = ldu4(v0); bv.u[1] = ldu4(v0 + 16);
    o0 = wmma_bf16(ap, bv, o0);
    const unsigned short* v1 = v0 + 16 * 32;
    bv.u[0] = ldu4(v1); bv.u[1] = ldu4(v1 + 16);
    o1 = wmma_bf16(ap, bv, o1);
    const unsigned short* v2 = v1 + 16 * 32;
    bv.u[0] = ldu4(v2); bv.u[1] = ldu4(v2 + 16);
    o2 = wmma_bf16(ap, bv, o2);
    const unsigned short* v3 = v2 + 16 * 32;
    bv.u[0] = ldu4(v3); bv.u[1] = ldu4(v3 + 16);
    o3 = wmma_bf16(ap, bv, o3);
  }

  if (lane < 16) sinv[w][lane] = 1.0f / ssum[w][lane];
#pragma unroll
  for (int r = 0; r < 8; ++r) {
    int l = l0 + r + half * 8;
    float inv = sinv[w][r + half * 8];
    size_t base = ((size_t)l * N_ + n) * E_ + h * HD_;
    aobf[base + lm]      = f2bf(o0[r] * inv);
    aobf[base + 16 + lm] = f2bf(o1[r] * inv);
    aobf[base + 32 + lm] = f2bf(o2[r] * inv);
    aobf[base + 48 + lm] = f2bf(o3[r] * inv);
  }
}

// out_proj + bias + residual -> fp32 y
__global__ void k_oproj(const unsigned short* __restrict__ aobf,
                        const unsigned short* __restrict__ w,
                        const float* __restrict__ bias,
                        const float* __restrict__ query,
                        float* __restrict__ yw) {
  const int tm = blockIdx.x;
  const int tn = blockIdx.y * blockDim.y + threadIdx.y;
  v8f c = gemm16x16<E_>(aobf, w, tm * 16, tn * 16);
  const int lane = threadIdx.x, half = lane >> 4, lm = lane & 15;
  const int col = tn * 16 + lm;
#pragma unroll
  for (int r = 0; r < 8; ++r) {
    int m = tm * 16 + r + half * 8;
    size_t idx = (size_t)m * E_ + col;
    yw[idx] = c[r] + bias[col] + query[idx];
  }
}

// LayerNorm over E per row
__global__ void k_ln(const float* __restrict__ yw,
                     const float* __restrict__ gamma,
                     const float* __restrict__ beta,
                     float* __restrict__ out) {
  const int row = blockIdx.x;
  const int tid = threadIdx.x;
  const float* yr = yw + (size_t)row * E_;
  float a = yr[tid], b = yr[tid + 256];
  __shared__ float s1[256], s2[256];
  s1[tid] = a + b;
  s2[tid] = a * a + b * b;
  __syncthreads();
  for (int off = 128; off > 0; off >>= 1) {
    if (tid < off) { s1[tid] += s1[tid + off]; s2[tid] += s2[tid + off]; }
    __syncthreads();
  }
  float mu = s1[0] * (1.0f / E_);
  float var = s2[0] * (1.0f / E_) - mu * mu;
  float rstd = rsqrtf(var + 1e-5f);
  out[(size_t)row * E_ + tid] = (a - mu) * rstd * gamma[tid] + beta[tid];
  out[(size_t)row * E_ + tid + 256] =
      (b - mu) * rstd * gamma[tid + 256] + beta[tid + 256];
}

// ---------------------------------------------------------------------------
extern "C" void kernel_launch(void* const* d_in, const int* in_sizes, int n_in,
                              void* d_out, int out_size, void* d_ws,
                              size_t ws_size, hipStream_t stream) {
  (void)in_sizes; (void)n_in; (void)out_size; (void)ws_size;
  const float* query = (const float*)d_in[0];
  const float* value = (const float*)d_in[1];
  const float* qpos  = (const float*)d_in[2];
  const float* vpos  = (const float*)d_in[3];
  const float* inW   = (const float*)d_in[4];
  const float* inB   = (const float*)d_in[5];
  const float* outW  = (const float*)d_in[6];
  const float* outB  = (const float*)d_in[7];
  const float* gamma = (const float*)d_in[8];
  const float* beta  = (const float*)d_in[9];

  unsigned short* wInBf  = (unsigned short*)d_ws;
  unsigned short* wOutBf = wInBf  + (size_t)3 * E_ * E_;
  unsigned short* qInBf  = wOutBf + (size_t)E_ * E_;
  unsigned short* vInBf  = qInBf  + (size_t)L_ * N_ * E_;
  unsigned short* qRot   = vInBf  + (size_t)S_ * N_ * E_;
  unsigned short* kRot   = qRot   + (size_t)N_ * H_ * L_ * HD_;
  unsigned short* vT     = kRot   + (size_t)N_ * H_ * S_ * HD_;
  unsigned short* aoBf   = vT     + (size_t)N_ * H_ * HD_ * S_;
  float* yW = (float*)(aoBf + (size_t)L_ * N_ * E_);

  k_convert<<<dim3(3 * E_ * E_ / 256), dim3(256), 0, stream>>>(inW, wInBf, 3 * E_ * E_);
  k_convert<<<dim3(E_ * E_ / 256), dim3(256), 0, stream>>>(outW, wOutBf, E_ * E_);
  k_convert<<<dim3(L_ * N_ * E_ / 256), dim3(256), 0, stream>>>(query, qInBf, L_ * N_ * E_);
  k_convert<<<dim3(S_ * N_ * E_ / 256), dim3(256), 0, stream>>>(value, vInBf, S_ * N_ * E_);

  k_qproj<<<dim3((L_ * N_) / 16, (E_ / 16) / 4), dim3(32, 4), 0, stream>>>(
      qInBf, wInBf, inB, qpos, qRot);
  k_kvproj<<<dim3((S_ * N_) / 16, (2 * E_ / 16) / 4), dim3(32, 4), 0, stream>>>(
      vInBf, wInBf + (size_t)E_ * E_, inB + E_, vpos, kRot, vT);

  k_attn<<<dim3(L_ / 64, N_ * H_), dim3(32, 4), 0, stream>>>(qRot, kRot, vT, aoBf);

  k_oproj<<<dim3((L_ * N_) / 16, (E_ / 16) / 4), dim3(32, 4), 0, stream>>>(
      aoBf, wOutBf, outB, query, yW);
  k_ln<<<dim3(L_ * N_), dim3(256), 0, stream>>>(yW, gamma, beta, (float*)d_out);
}
